// TransformerBlock_10307921511168
// MI455X (gfx1250) — matrix operand order repaired and checked by reading, not compiled, not run on hardware
//
#include <hip/hip_runtime.h>

// ---------------------------------------------------------------------------
// Transformer block for MI455X (gfx1250, wave32, WMMA).
// B=4, T=2048, D=768, H=12, HD=64.  All matmuls on v_wmma_f32_16x16x32_bf16.
// GEMM tiles staged with GLOBAL_LOAD_ASYNC_TO_LDS_B128 + double buffering.
// ---------------------------------------------------------------------------

#define TT   2048
#define DD   768
#define HH   12
#define HDD  64
#define BT   8192           // B*T
#define QKVN 2304           // 3*D
#define FFD  3072           // 4*D

typedef __attribute__((ext_vector_type(16))) __bf16 v16bf;
typedef __attribute__((ext_vector_type(8)))  float  v8f;

union FragBF {                 // one WMMA 16-bit A or B operand (8 VGPRs)
    v16bf v;
    unsigned short u[16];
    uint4 q[2];
};

__device__ __forceinline__ unsigned short f2bf(float f) {
    unsigned int x = __float_as_uint(f);
    x += 0x7fffu + ((x >> 16) & 1u);          // round-to-nearest-even
    return (unsigned short)(x >> 16);
}

__device__ __forceinline__ v8f wmma_bf16(v16bf a, v16bf b, v8f c) {
    return __builtin_amdgcn_wmma_f32_16x16x32_bf16(
        /*neg_a=*/false, a, /*neg_b=*/false, b,
        /*c_mod=*/(short)0, c, /*reuse_a=*/false, /*reuse_b=*/false);
}

// --- CDNA5 async global->LDS staging (ASYNCcnt path), with safe fallback ----
#if defined(__gfx1250__) && __has_builtin(__builtin_amdgcn_global_load_async_to_lds_b128)
#define ASYNC_LDS 1
#endif

#ifdef ASYNC_LDS
typedef int b128_t __attribute__((vector_size(16)));
typedef __attribute__((address_space(1))) b128_t* gp128_t;   // global src
typedef __attribute__((address_space(3))) b128_t* lp128_t;   // LDS dst
#endif

__device__ __forceinline__ void stage16(const unsigned short* g, unsigned short* l) {
#ifdef ASYNC_LDS
    __builtin_amdgcn_global_load_async_to_lds_b128(
        (gp128_t)(void*)g, (lp128_t)l, /*imm offset=*/0, /*cpol=*/0);
#else
    *(uint4*)l = *(const uint4*)g;
#endif
}

__device__ __forceinline__ void async_wait0() {
#ifdef ASYNC_LDS
#if __has_builtin(__builtin_amdgcn_s_wait_asynccnt)
    __builtin_amdgcn_s_wait_asynccnt(0);
#else
    asm volatile("s_wait_asynccnt 0" ::: "memory");
#endif
#endif
}

// ---------------------------------------------------------------------------
// Weight conversion kernels: fp32 -> bf16, stored transposed [N][K] so WMMA
// B-fragments are two 16B runs along K per lane.
// ---------------------------------------------------------------------------

// Fused QKV weight: out[np*768 + k], np in [0,2304): 0-767=Q, 768-1535=K, rest=V.
// Source is [H, D, HD]: element (h, k, c) at h*D*HD + k*HD + c;  np = h*64+c.
__global__ __launch_bounds__(256) void conv_qkvT(const float* __restrict__ Wq,
                                                 const float* __restrict__ Wk,
                                                 const float* __restrict__ Wv,
                                                 unsigned short* __restrict__ out) {
    int i = blockIdx.x * 256 + threadIdx.x;
    if (i >= QKVN * DD) return;
    int k  = i % DD;
    int np = i / DD;
    const float* W = Wq;
    int base = np;
    if (np >= 2 * DD)      { W = Wv; base = np - 2 * DD; }
    else if (np >= DD)     { W = Wk; base = np - DD; }
    int h = base / HDD, c = base % HDD;
    out[i] = f2bf(W[(size_t)h * DD * HDD + (size_t)k * HDD + c]);
}

__global__ __launch_bounds__(256) void conv_qkv_bias(const float* __restrict__ bq,
                                                     const float* __restrict__ bk,
                                                     const float* __restrict__ bv,
                                                     float* __restrict__ out) {
    int i = blockIdx.x * 256 + threadIdx.x;
    if (i >= QKVN) return;
    out[i] = (i < DD) ? bq[i] : (i < 2 * DD) ? bk[i - DD] : bv[i - 2 * DD];
}

// Generic transpose-convert: src is [rows(K), cols(N)] fp32; out[n*rows + k] bf16.
__global__ __launch_bounds__(256) void convT(const float* __restrict__ in,
                                             unsigned short* __restrict__ out,
                                             int rows, int cols) {
    size_t i = (size_t)blockIdx.x * 256 + threadIdx.x;
    if (i >= (size_t)rows * cols) return;
    int k = (int)(i % rows);
    int n = (int)(i / rows);
    out[i] = f2bf(in[(size_t)k * cols + n]);
}

// ---------------------------------------------------------------------------
// LayerNorm (one block per token row), fp32 in -> bf16 out.
// ---------------------------------------------------------------------------
__global__ __launch_bounds__(256) void ln_kernel(const float* __restrict__ x,
                                                 const float* __restrict__ g,
                                                 const float* __restrict__ be,
                                                 unsigned short* __restrict__ out) {
    __shared__ float red[256];
    int row = blockIdx.x, tid = threadIdx.x;
    const float* xr = x + (size_t)row * DD;
    float v0 = xr[tid], v1 = xr[tid + 256], v2 = xr[tid + 512];
    red[tid] = v0 + v1 + v2;
    __syncthreads();
    for (int off = 128; off > 0; off >>= 1) {
        if (tid < off) red[tid] += red[tid + off];
        __syncthreads();
    }
    float mu = red[0] * (1.0f / DD);
    __syncthreads();
    float d0 = v0 - mu, d1 = v1 - mu, d2 = v2 - mu;
    red[tid] = d0 * d0 + d1 * d1 + d2 * d2;
    __syncthreads();
    for (int off = 128; off > 0; off >>= 1) {
        if (tid < off) red[tid] += red[tid + off];
        __syncthreads();
    }
    float rstd = rsqrtf(red[0] * (1.0f / DD) + 1e-5f);
    unsigned short* orow = out + (size_t)row * DD;
    orow[tid]       = f2bf(d0 * rstd * g[tid]       + be[tid]);
    orow[tid + 256] = f2bf(d1 * rstd * g[tid + 256] + be[tid + 256]);
    orow[tid + 512] = f2bf(d2 * rstd * g[tid + 512] + be[tid + 512]);
}

// ---------------------------------------------------------------------------
// WMMA GEMM:  C[M,N] = A[M,K](bf16) * Bt[N,K](bf16)^T + bias  (+ epilogue)
// Block tile 128x128, 8 waves (4 rows x 2 cols), wave tile 32x64 (2x4 frags),
// K-step 32.  Tiles double-buffered in LDS, filled by async global->LDS.
// EPI: 0 = store bf16;  1 = store fp32 + residual;  2 = GELU -> bf16.
// ---------------------------------------------------------------------------
template <int EPI>
__global__ __launch_bounds__(256) void gemm_kernel(const unsigned short* __restrict__ A,
                                                   const unsigned short* __restrict__ Bt,
                                                   const float* __restrict__ bias,
                                                   const float* __restrict__ resid,
                                                   void* __restrict__ outv,
                                                   int M, int N, int K) {
    __shared__ unsigned short As[2][128][32];   // 16 KB
    __shared__ unsigned short Bs[2][128][32];   // 16 KB

    const int tid  = threadIdx.x;
    const int w    = tid >> 5;
    const int lane = tid & 31;
    const int hf   = lane >> 4;     // 0/1: 16-lane half
    const int ln   = lane & 15;
    const int wm   = w & 3;         // wave row in 4x2 wave grid (32 rows each)
    const int wn   = w >> 2;        // wave col (64 cols each)
    const int bm   = blockIdx.y * 128;
    const int bn   = blockIdx.x * 128;

    // Staging assignment: 512 x b128 per tile, 2 per thread per tile.
    const int srow = tid >> 2;            // 0..63 (+64 on second pass)
    const int sc8  = (tid & 3) * 8;       // 0/8/16/24

    auto stage_tiles = [&](int buf, int kt) {
#pragma unroll
        for (int e = 0; e < 2; ++e) {
            int row = srow + e * 64;
            stage16(A + (size_t)(bm + row) * K + kt * 32 + sc8, &As[buf][row][sc8]);
            stage16(Bt + (size_t)(bn + row) * K + kt * 32 + sc8, &Bs[buf][row][sc8]);
        }
    };

    v8f c[2][4];
#pragma unroll
    for (int i = 0; i < 2; ++i)
#pragma unroll
        for (int j = 0; j < 4; ++j)
#pragma unroll
            for (int e = 0; e < 8; ++e) c[i][j][e] = 0.0f;

    const int nk = K / 32;
    stage_tiles(0, 0);
    int cur = 0;

    for (int kt = 0; kt < nk; ++kt) {
        async_wait0();          // this wave's async fills landed in LDS
        __syncthreads();        // everyone's fills landed; prev reads done
        if (kt + 1 < nk) stage_tiles(cur ^ 1, kt + 1);   // prefetch next tile

        // A fragments: 16-bit A 16x32 layout (lane r=M; halves K 0-7/16-23 | 8-15/24-31).
        FragBF a[2];
#pragma unroll
        for (int i = 0; i < 2; ++i) {
            int row = wm * 32 + i * 16 + ln;
            int k0  = hf * 8;
            a[i].q[0] = *(const uint4*)&As[cur][row][k0];
            a[i].q[1] = *(const uint4*)&As[cur][row][k0 + 16];
        }
        // B fragments: lane n=N column; halves K 0-7/16-23 | 8-15/24-31.
#pragma unroll
        for (int j = 0; j < 4; ++j) {
            FragBF bfr;
            int row = wn * 64 + j * 16 + ln;
            int kk  = hf * 8;
            bfr.q[0] = *(const uint4*)&Bs[cur][row][kk];
            bfr.q[1] = *(const uint4*)&Bs[cur][row][kk + 16];
#pragma unroll
            for (int i = 0; i < 2; ++i)
                c[i][j] = wmma_bf16(a[i].v, bfr.v, c[i][j]);
        }
        cur ^= 1;
    }

    // Epilogue. C layout: n = ln, m = r + 8*hf within each 16x16 frag.
#pragma unroll
    for (int i = 0; i < 2; ++i)
#pragma unroll
        for (int j = 0; j < 4; ++j)
#pragma unroll
            for (int r = 0; r < 8; ++r) {
                int m = bm + wm * 32 + i * 16 + hf * 8 + r;
                int n = bn + wn * 64 + j * 16 + ln;
                float val = c[i][j][r] + bias[n];
                size_t off = (size_t)m * N + n;
                if (EPI == 0) {
                    ((unsigned short*)outv)[off] = f2bf(val);
                } else if (EPI == 1) {
                    ((float*)outv)[off] = val + resid[off];
                } else {  // exact GELU
                    float ge = 0.5f * val * (1.0f + erff(val * 0.70710678118f));
                    ((unsigned short*)outv)[off] = f2bf(ge);
                }
            }
}

// ---------------------------------------------------------------------------
// Flash attention (causal, online softmax).  4 waves / block, each wave owns
// 16 query rows; streams key/value tiles of 32.  Both matmuls are WMMA bf16.
// qkv rows: [BT, 2304] bf16, cols 0-767 Q(h-major), 768-1535 K, 1536-2303 V.
// K tile staged via async global->LDS; V tile transposed manually.
// ---------------------------------------------------------------------------
__global__ __launch_bounds__(128) void attn_kernel(const unsigned short* __restrict__ qkv,
                                                   unsigned short* __restrict__ obf) {
    const int qt = blockIdx.x, h = blockIdx.y, b = blockIdx.z;
    const int tid = threadIdx.x, w = tid >> 5, lane = tid & 31;
    const int hf = lane >> 4, ln = lane & 15;

    __shared__ unsigned short Ks[32][64];    // [key][d]   (serves as Bt for QK^T)
    __shared__ unsigned short Vt[64][32];    // [d][key]   (serves as Bt for P*V)
    __shared__ unsigned short Ps[4][16][32]; // per-wave P staging (C-layout -> A-layout)

    const int qbase = qt * 64 + w * 16;

    // Q A-fragments for this wave's 16 rows (d split in two 32-chunks).
    FragBF aq[2];
    {
        const unsigned short* qrow =
            qkv + (size_t)(b * TT + qbase + ln) * QKVN + h * HDD;
#pragma unroll
        for (int dd = 0; dd < 2; ++dd) {
            int k0 = dd * 32 + hf * 8;
            aq[dd].q[0] = *(const uint4*)(qrow + k0);
            aq[dd].q[1] = *(const uint4*)(qrow + k0 + 16);
        }
    }

    float l[8], mx[8];
    v8f oacc[4];
#pragma unroll
    for (int r = 0; r < 8; ++r) { l[r] = 0.0f; mx[r] = -1e30f; }
#pragma unroll
    for (int dn = 0; dn < 4; ++dn)
#pragma unroll
        for (int e = 0; e < 8; ++e) oacc[dn][e] = 0.0f;

    const float scale = 0.125f;      // 1/sqrt(64)
    const int ntiles = qt * 2 + 2;   // causal: keys < (qt+1)*64

    for (int kt = 0; kt < ntiles; ++kt) {
        __syncthreads();
        // Stage K tile 32x64 via async (256 b128, 2/thread).
#pragma unroll
        for (int e = 0; e < 2; ++e) {
            int idx = tid + e * 128;
            int row = idx >> 3, c8 = (idx & 7) * 8;
            stage16(qkv + (size_t)(b * TT + kt * 32 + row) * QKVN + DD + h * HDD + c8,
                    &Ks[row][c8]);
        }
        // Stage V tile transposed into Vt[d][key] (manual scatter).
#pragma unroll
        for (int e = 0; e < 2; ++e) {
            int lin = tid * 2 + e;          // 0..255
            int key = lin >> 3, d8 = (lin & 7) * 8;
            uint4 t = *(const uint4*)(
                qkv + (size_t)(b * TT + kt * 32 + key) * QKVN + 2 * DD + h * HDD + d8);
            const unsigned short* ts = (const unsigned short*)&t;
#pragma unroll
            for (int ii = 0; ii < 8; ++ii) Vt[d8 + ii][key] = ts[ii];
        }
        async_wait0();
        __syncthreads();

        // S = Q * K^T  (16 q-rows x 32 keys), contraction over HD=64 in 2 steps.
        v8f s[2];
#pragma unroll
        for (int j = 0; j < 2; ++j)
#pragma unroll
            for (int e = 0; e < 8; ++e) s[j][e] = 0.0f;
#pragma unroll
        for (int dd = 0; dd < 2; ++dd) {
            int d0 = dd * 32;
#pragma unroll
            for (int j = 0; j < 2; ++j) {
                FragBF bk;
                int row = j * 16 + ln;           // key column (N)
                int kk  = d0 + hf * 8;           // contraction chunk
                bk.q[0] = *(const uint4*)&Ks[row][kk];
                bk.q[1] = *(const uint4*)&Ks[row][kk + 16];
                s[j] = wmma_bf16(aq[dd].v, bk.v, s[j]);
            }
        }

        // Online softmax: rows m = hf*8 + r, key n = kt*32 + j*16 + ln.
        float pr[2][8];
#pragma unroll
        for (int r = 0; r < 8; ++r) {
            int m  = hf * 8 + r;
            int qg = qbase + m;
            float rowm = -1e30f;
#pragma unroll
            for (int j = 0; j < 2; ++j) {
                float v = s[j][r] * scale;
                int kg = kt * 32 + j * 16 + ln;
                if (kg > qg) v = -1e30f;
                pr[j][r] = v;
                rowm = fmaxf(rowm, v);
            }
#pragma unroll
            for (int off = 1; off < 16; off <<= 1)
                rowm = fmaxf(rowm, __shfl_xor(rowm, off, 32));
            float mnew  = fmaxf(mx[r], rowm);
            float alpha = __expf(mx[r] - mnew);
            mx[r] = mnew;
            float rs = 0.0f;
#pragma unroll
            for (int j = 0; j < 2; ++j) {
                pr[j][r] = __expf(pr[j][r] - mnew);
                rs += pr[j][r];
            }
#pragma unroll
            for (int off = 1; off < 16; off <<= 1)
                rs += __shfl_xor(rs, off, 32);
            l[r] = l[r] * alpha + rs;
#pragma unroll
            for (int dn = 0; dn < 4; ++dn) oacc[dn][r] *= alpha;
            // Stage P (C-layout -> LDS) for A-fragment reload.
#pragma unroll
            for (int j = 0; j < 2; ++j)
                Ps[w][m][j * 16 + ln] = f2bf(pr[j][r]);
        }
        asm volatile("s_wait_dscnt 0" ::: "memory");  // wave-private LDS RAW

        // Reload P in A layout (16x32).
        FragBF ap;
        {
            const unsigned short* prow = &Ps[w][ln][0];
            int k0 = hf * 8;
            ap.q[0] = *(const uint4*)(prow + k0);
            ap.q[1] = *(const uint4*)(prow + k0 + 16);
        }
        // O += P * V  (Bt = Vt[d][key]).
#pragma unroll
        for (int dn = 0; dn < 4; ++dn) {
            FragBF bv;
            int row = dn * 16 + ln;  // output d column (N)
            int kk  = hf * 8;        // contraction over 32 keys
            bv.q[0] = *(const uint4*)&Vt[row][kk];
            bv.q[1] = *(const uint4*)&Vt[row][kk + 16];
            oacc[dn] = wmma_bf16(ap.v, bv.v, oacc[dn]);
        }
    }

    // Normalize and write out: obf[bt, h*64 + d] (heads concatenated h-major).
#pragma unroll
    for (int r = 0; r < 8; ++r) {
        int m = hf * 8 + r;
        size_t row = (size_t)(b * TT + qbase + m) * DD + h * HDD;
        float inv = 1.0f / l[r];
#pragma unroll
        for (int dn = 0; dn < 4; ++dn)
            obf[row + dn * 16 + ln] = f2bf(oacc[dn][r] * inv);
    }
}

// ---------------------------------------------------------------------------
// Host launcher.
// ---------------------------------------------------------------------------
extern "C" void kernel_launch(void* const* d_in, const int* in_sizes, int n_in,
                              void* d_out, int out_size, void* d_ws, size_t ws_size,
                              hipStream_t stream) {
    (void)in_sizes; (void)n_in; (void)out_size; (void)ws_size;

    const float* x     = (const float*)d_in[0];
    const float* ln1_g = (const float*)d_in[1];
    const float* ln1_b = (const float*)d_in[2];
    const float* Wq    = (const float*)d_in[3];
    const float* bq    = (const float*)d_in[4];
    const float* Wk    = (const float*)d_in[5];
    const float* bk    = (const float*)d_in[6];
    const float* Wv    = (const float*)d_in[7];
    const float* bv    = (const float*)d_in[8];
    const float* Wo    = (const float*)d_in[9];
    const float* bo    = (const float*)d_in[10];
    const float* ln2_g = (const float*)d_in[11];
    const float* ln2_b = (const float*)d_in[12];
    const float* W1    = (const float*)d_in[13];
    const float* b1    = (const float*)d_in[14];
    const float* W2    = (const float*)d_in[15];
    const float* b2    = (const float*)d_in[16];

    // Workspace carve-up (256B aligned).
    size_t off = 0;
    auto nxt = [&](size_t bytes) -> char* {
        char* p = (char*)d_ws + off;
        off += (bytes + 255) & ~(size_t)255;
        return p;
    };
    unsigned short* qkvT = (unsigned short*)nxt((size_t)QKVN * DD * 2);
    float*          qkvb = (float*)         nxt((size_t)QKVN * 4);
    unsigned short* WoT  = (unsigned short*)nxt((size_t)DD * DD * 2);
    unsigned short* W1T  = (unsigned short*)nxt((size_t)FFD * DD * 2);
    unsigned short* W2T  = (unsigned short*)nxt((size_t)DD * FFD * 2);
    unsigned short* hbf  = (unsigned short*)nxt((size_t)BT * DD * 2);
    unsigned short* qkv  = (unsigned short*)nxt((size_t)BT * QKVN * 2);
    unsigned short* obf  = (unsigned short*)nxt((size_t)BT * DD * 2);
    float*          x1   = (float*)         nxt((size_t)BT * DD * 4);
    unsigned short* h2   = (unsigned short*)nxt((size_t)BT * DD * 2);
    unsigned short* ffb  = (unsigned short*)nxt((size_t)BT * FFD * 2);

    // 1. Weight prep (bf16, transposed [N,K]).
    conv_qkvT<<<(QKVN * DD + 255) / 256, 256, 0, stream>>>(Wq, Wk, Wv, qkvT);
    conv_qkv_bias<<<(QKVN + 255) / 256, 256, 0, stream>>>(bq, bk, bv, qkvb);
    convT<<<(DD * DD + 255) / 256, 256, 0, stream>>>(Wo, WoT, DD, DD);
    convT<<<(DD * FFD + 255) / 256, 256, 0, stream>>>(W1, W1T, DD, FFD);
    convT<<<(FFD * DD + 255) / 256, 256, 0, stream>>>(W2, W2T, FFD, DD);

    // 2. LN1 -> hbf.
    ln_kernel<<<BT, 256, 0, stream>>>(x, ln1_g, ln1_b, hbf);

    // 3. Fused QKV projection: [8192,768] x [768,2304] -> qkv bf16.
    gemm_kernel<0><<<dim3(QKVN / 128, BT / 128), 256, 0, stream>>>(
        hbf, qkvT, qkvb, nullptr, qkv, BT, QKVN, DD);

    // 4. Flash attention.
    attn_kernel<<<dim3(TT / 64, HH, 4), 128, 0, stream>>>(qkv, obf);

    // 5. Output projection + residual: x1 = x + o @ Wo + bo.
    gemm_kernel<1><<<dim3(DD / 128, BT / 128), 256, 0, stream>>>(
        obf, WoT, bo, x, x1, BT, DD, DD);

    // 6. LN2 -> h2.
    ln_kernel<<<BT, 256, 0, stream>>>(x1, ln2_g, ln2_b, h2);

    // 7. FFN1 + exact GELU: ff = gelu(h2 @ W1 + b1).
    gemm_kernel<2><<<dim3(FFD / 128, BT / 128), 256, 0, stream>>>(
        h2, W1T, b1, nullptr, ffb, BT, FFD, DD);

    // 8. FFN2 + residual: out = x1 + ff @ W2 + b2.
    gemm_kernel<1><<<dim3(DD / 128, BT / 128), 256, 0, stream>>>(
        ffb, W2T, b2, x1, (float*)d_out, BT, DD, FFD);
}
